// LaplacianLoss_60086592471431
// MI455X (gfx1250) — compile-verified, matrix operand order checked by reading, not run on hardware
//
#include <hip/hip_runtime.h>

typedef __attribute__((ext_vector_type(2)))  float    v2f;
typedef __attribute__((ext_vector_type(8)))  float    v8f;
typedef __attribute__((ext_vector_type(16))) _Float16 v16h;

__device__ __forceinline__ float wave_reduce_sum(float v) {
#pragma unroll
  for (int off = 16; off > 0; off >>= 1)
    v += __shfl_xor(v, off, 32);
  return v;
}

// ---------------------------------------------------------------------------
// Kernel 1: per-row squared norms. One wave32 per row; lane l loads float4 at
// column 4*l (128 floats = 32 lanes * 4), squares, wave-reduces.
// ---------------------------------------------------------------------------
__global__ void lap_xsq_kernel(const float* __restrict__ F,
                               float* __restrict__ xsq, int N) {
  const int lane = (int)(threadIdx.x & 31u);
  const int row  = (int)((blockIdx.x * blockDim.x + threadIdx.x) >> 5);
  if (row >= N) return;
  const float4 v = ((const float4*)(F + (size_t)row * 128))[lane];
  float s = v.x * v.x + v.y * v.y + v.z * v.z + v.w * v.w;
  s = wave_reduce_sum(s);
  if (lane == 0) xsq[row] = s;
}

// ---------------------------------------------------------------------------
// Kernel 2: 16 edges per wave via V_WMMA_F32_16X16X4_F32.
//   A (16x4 f32): lane l<16  -> row i[e0+l],  K = {4k+0, 4k+1}
//                 lane l>=16 -> row i[e0+l-16], K = {4k+2, 4k+3}
//   B ( 4x16 f32): same pattern over rows j[...] (columns across lanes).
//   D = A*B accumulated over 32 K-steps; only the diagonal D[m][m] is used:
//     m<8  -> VGPR m,   lane m
//     m>=8 -> VGPR m-8, lane m+16  (i.e. lanes 24..31 hold c[lane-24])
// ---------------------------------------------------------------------------
__global__ void lap_edge_kernel(const float* __restrict__ F,
                                const float* __restrict__ xsq,
                                const long long* __restrict__ idx, // [2*E]
                                float* __restrict__ accum,
                                long long E) {
  const int lane = (int)(threadIdx.x & 31u);
  const long long wave =
      (long long)blockIdx.x * (blockDim.x >> 5) + (threadIdx.x >> 5);
  const long long e0 = wave << 4;                 // 16 edges per wave
  if (e0 >= E) return;

  const int slot = lane & 15;                     // edge slot 0..15
  const int kh   = lane >> 4;                     // 0: K={0,1}, 1: K={2,3}
  long long e = e0 + slot;
  const bool ev = (e < E);
  if (!ev) e = E - 1;                             // clamp; keep EXEC all-ones
  const long long ri = idx[e];
  const long long rj = idx[E + e];

  const float* __restrict__ Abase = F + ri * 128;
  const float* __restrict__ Bbase = F + rj * 128;

  v8f c = {0.f, 0.f, 0.f, 0.f, 0.f, 0.f, 0.f, 0.f};

#if __has_builtin(__builtin_amdgcn_wmma_f32_16x16x4_f32)
  const float* __restrict__ Ap = Abase + kh * 2;
  const float* __restrict__ Bp = Bbase + kh * 2;
#pragma unroll 8
  for (int kb = 0; kb < 32; ++kb) {               // K = 4*kb .. 4*kb+3
    v2f a = *(const v2f*)(Ap + kb * 4);
    v2f b = *(const v2f*)(Bp + kb * 4);
    c = __builtin_amdgcn_wmma_f32_16x16x4_f32(
        /*neg_a=*/false, a, /*neg_b=*/false, b,
        /*c_mod=*/(short)0, c, /*reuse_a=*/false, /*reuse_b=*/false);
  }
#else
  // Fallback: f16 WMMA path (codegen-confirmed builtin), reduced precision.
#pragma unroll
  for (int kb = 0; kb < 4; ++kb) {                // K = 32*kb .. 32*kb+31
    v16h a, b;
#pragma unroll
    for (int t = 0; t < 16; ++t) {
      const int vg = t >> 1, p = t & 1;
      const int k = ((vg < 4) ? (vg * 2) : (16 + (vg - 4) * 2)) + p + kh * 8;
      a[t] = (_Float16)Abase[kb * 32 + k];
      b[t] = (_Float16)Bbase[kb * 32 + k];
    }
    c = __builtin_amdgcn_wmma_f32_16x16x32_f16(
        false, a, false, b, (short)0, c, false, false);
  }
#endif

  // --- diagonal extraction ---
  int msel = -1;
  if (lane < 8)        msel = lane;               // edge m = lane
  else if (lane >= 24) msel = lane - 24;          // edge m = lane - 16
  float diag = 0.f;
#pragma unroll
  for (int r = 0; r < 8; ++r) diag = (msel == r) ? c[r] : diag;
  const long long em = (lane < 8) ? (e0 + lane) : (e0 + lane - 16);
  if (msel >= 0 && em >= E) diag = 0.f;           // mask tail edges

  // --- x1 + x2 (lanes 0..15 own one edge each) ---
  float xs = 0.f;
  if (lane < 16 && ev) xs = xsq[ri] + xsq[rj];

  const float contrib = wave_reduce_sum(xs - 2.0f * diag);
  if (lane == 0) atomicAdd(accum, contrib);
}

__global__ void lap_zero_kernel(float* p) {
  if (threadIdx.x == 0) p[0] = 0.0f;
}

__global__ void lap_finalize_kernel(const float* __restrict__ accum,
                                    float* __restrict__ out, long long E) {
  if (threadIdx.x == 0) out[0] = accum[0] / (float)E;
}

extern "C" void kernel_launch(void* const* d_in, const int* in_sizes, int n_in,
                              void* d_out, int out_size, void* d_ws,
                              size_t ws_size, hipStream_t stream) {
  const float*     F   = (const float*)d_in[0];
  const long long* idx = (const long long*)d_in[1];   // int64 indices [2, E]
  const long long  N   = (long long)in_sizes[0] / 128;
  const long long  E   = (long long)in_sizes[1] / 2;

  float* accum = (float*)d_ws;                         // 1 float
  float* xsq   = (float*)((char*)d_ws + 256);          // N floats

  lap_zero_kernel<<<dim3(1), dim3(64), 0, stream>>>(accum);

  {  // squared norms: one wave per row, 8 waves per block
    const long long blocks = (N + 7) / 8;
    lap_xsq_kernel<<<dim3((unsigned)blocks), dim3(256), 0, stream>>>(
        F, xsq, (int)N);
  }

  {  // edge tiles: 16 edges per wave, 8 waves per block
    const long long tiles  = (E + 15) / 16;
    const long long blocks = (tiles + 7) / 8;
    lap_edge_kernel<<<dim3((unsigned)blocks), dim3(256), 0, stream>>>(
        F, xsq, idx, accum, E);
  }

  lap_finalize_kernel<<<dim3(1), dim3(64), 0, stream>>>(accum, (float*)d_out, E);
}
